// GCCONV_61083024883832
// MI455X (gfx1250) — matrix operand order repaired and checked by reading, not compiled, not run on hardware
//
#include <hip/hip_runtime.h>
#include <hip/hip_bf16.h>

typedef __attribute__((ext_vector_type(16))) _Float16 v16h;
typedef __attribute__((ext_vector_type(8)))  float    v8f;
typedef __attribute__((ext_vector_type(4)))  unsigned u32x4;

#define IN_FEATS  256
#define OUT_FEATS 128
#define ROWS_PER_BLOCK 16
// A tile pitch in halfs: 256 + 8 keeps rows 16B-aligned (264*2 = 528 = 33*16)
#define A_PITCH 264

// ---------------- zero ----------------
__global__ void GC_zero_f32(float* __restrict__ p, long n) {
  long i = (long)blockIdx.x * blockDim.x + threadIdx.x;
  long stride = (long)gridDim.x * blockDim.x;
  for (; i < n; i += stride) p[i] = 0.0f;
}

// ---------------- degree histograms ----------------
__global__ void GC_degrees(const int* __restrict__ src, const int* __restrict__ dst,
                           float* __restrict__ outdeg, float* __restrict__ indeg, int E) {
  int e = blockIdx.x * blockDim.x + threadIdx.x;
  if (e < E) {
    atomicAdd(&outdeg[src[e]], 1.0f);
    atomicAdd(&indeg[dst[e]], 1.0f);
  }
}

// ---------------- one-time W transpose+convert: Wt[n][k] = (f16)W[k][n] ----------------
__global__ void GC_prepW(const float* __restrict__ W, _Float16* __restrict__ Wt) {
  int idx = blockIdx.x * blockDim.x + threadIdx.x;   // 0..32767
  if (idx < IN_FEATS * OUT_FEATS) {
    int k = idx & (IN_FEATS - 1);
    int n = idx >> 8;
    Wt[idx] = (_Float16)W[(long)k * OUT_FEATS + n];
  }
}

// ---------------- WMMA GEMM + tanh ----------------
// h[r,n] = tanh( sum_k feat[r,k]*rsqrt(max(outdeg[r],1)) * W[k,n] )
// 256 threads = 8 waves; 16 rows/block; wave w owns column tile [16w,16w+16).
// A (16x256 fp16, scaled) staged once in LDS; B fragments streamed from L2-hot Wt.
__global__ __launch_bounds__(256) void GC_gemm_tanh(
    const float* __restrict__ feat, const _Float16* __restrict__ Wt,
    const float* __restrict__ outdeg, float* __restrict__ h, int Nn) {
  __shared__ _Float16 sA[ROWS_PER_BLOCK * A_PITCH];
  __shared__ float    sScale[ROWS_PER_BLOCK];

  const int tid  = threadIdx.x;
  const int wave = tid >> 5;
  const int lane = tid & 31;
  const int r0   = blockIdx.x * ROWS_PER_BLOCK;

  if (tid < ROWS_PER_BLOCK) {
    int r = r0 + tid;
    float d = (r < Nn) ? outdeg[r] : 1.0f;
    sScale[tid] = rsqrtf(fmaxf(d, 1.0f));
  }
  __syncthreads();

  // stage full A tile: 16 rows x 256 k = 1024 float4-chunks, 4 per thread
  #pragma unroll
  for (int it = 0; it < 4; ++it) {
    int c   = tid + it * 256;          // chunk 0..1023
    int row = c >> 6;                  // 64 chunks per row
    int kq  = (c & 63) * 4;
    int rr  = r0 + row;
    int r   = (rr < Nn) ? rr : (Nn - 1);   // clamp: no divergent branch
    float4 f = *(const float4*)&feat[(long)r * IN_FEATS + kq];
    float  s = sScale[row];
    union { _Float16 hh[4]; uint2 q; } pk;
    pk.hh[0] = (_Float16)(f.x * s);
    pk.hh[1] = (_Float16)(f.y * s);
    pk.hh[2] = (_Float16)(f.z * s);
    pk.hh[3] = (_Float16)(f.w * s);
    *(uint2*)&sA[row * A_PITCH + kq] = pk.q;
  }
  __syncthreads();

  const int m  = lane & 15;
  const int hi = lane >> 4;
  const int n  = 16 * wave + (lane & 15);
  const _Float16* __restrict__ brow = Wt + (long)n * IN_FEATS + 8 * hi;
  const int abase = m * A_PITCH + 8 * hi;   // halfs

  v8f c = {};
  #pragma unroll
  for (int kk = 0; kk < 8; ++kk) {
    const int k0 = kk * 32;
    // A fragment: two contiguous 16B runs -> 2x ds_load_b128
    union { v16h v; u32x4 q[2]; } a;
    a.q[0] = *(const u32x4*)&sA[abase + k0];        // pairs K = k0+{0..7}+8hi
    a.q[1] = *(const u32x4*)&sA[abase + k0 + 16];   // pairs K = k0+16+{0..7}+8hi
    // B fragment: two 8-half runs of row n (K = k0+{0..7}+8hi, k0+16+{0..7}+8hi) -> 2x global_load_b128 (L2-hot)
    union { v16h v; u32x4 q[2]; } b;
    b.q[0] = *(const u32x4*)&brow[k0];
    b.q[1] = *(const u32x4*)&brow[k0 + 16];

    c = __builtin_amdgcn_wmma_f32_16x16x32_f16(
        false, a.v, false, b.v, (short)0, c, false, false);
  }

  // store D with tanh: VGPR i -> M = i + 8*hi, N = n
  {
    int mrow = 8 * hi;
    #pragma unroll
    for (int i = 0; i < 8; ++i) {
      int r = r0 + mrow + i;
      if (r < Nn) h[(long)r * OUT_FEATS + n] = tanhf(c[i]);
    }
  }
}

// ---------------- edge scatter: out[dst] += h[src] * ew ----------------
// one wave per edge; each lane handles 4 consecutive feats (32*4 = 128)
__global__ __launch_bounds__(256) void GC_scatter(
    const float* __restrict__ h, const float* __restrict__ ew,
    const int* __restrict__ src, const int* __restrict__ dst,
    float* __restrict__ out, int E) {
  int lane = threadIdx.x & 31;
  int gw = blockIdx.x * (blockDim.x >> 5) + (threadIdx.x >> 5);
  int nw = gridDim.x * (blockDim.x >> 5);
  for (int e = gw; e < E; e += nw) {
    int s = src[e];
    int d = dst[e];
    float w = ew[e];
    const float4* hp = reinterpret_cast<const float4*>(h + (long)s * OUT_FEATS);
    float4 v = hp[lane];
    float* op = out + (long)d * OUT_FEATS + lane * 4;
    atomicAdd(op + 0, v.x * w);
    atomicAdd(op + 1, v.y * w);
    atomicAdd(op + 2, v.z * w);
    atomicAdd(op + 3, v.w * w);
  }
}

// ---------------- post-scale by in_deg^-1/2 ----------------
__global__ void GC_scale_in(float* __restrict__ out, const float* __restrict__ indeg, long total) {
  long i = (long)blockIdx.x * blockDim.x + threadIdx.x;
  if (i < total) {
    int node = (int)(i >> 7);   // /128
    out[i] *= rsqrtf(fmaxf(indeg[node], 1.0f));
  }
}

extern "C" void kernel_launch(void* const* d_in, const int* in_sizes, int n_in,
                              void* d_out, int out_size, void* d_ws, size_t ws_size,
                              hipStream_t stream) {
  const float* feat   = (const float*)d_in[0];  // [N,256]
  const float* weight = (const float*)d_in[1];  // [256,128]
  const float* ew     = (const float*)d_in[2];  // [E]
  const int*   src    = (const int*)d_in[3];    // [E]
  const int*   dst    = (const int*)d_in[4];    // [E]
  float* out = (float*)d_out;

  const int Nn = in_sizes[0] / IN_FEATS;
  const int E  = in_sizes[2];

  float*    wsf    = (float*)d_ws;
  float*    outdeg = wsf;                         // [N]
  float*    indeg  = wsf + Nn;                    // [N]
  _Float16* Wt     = (_Float16*)(wsf + 2L * Nn);  // [128,256] fp16 = 64KB
  float*    h      = wsf + 2L * Nn + (IN_FEATS * OUT_FEATS / 2);  // [N,128]

  // zero degrees (contiguous 2N) and output
  {
    long n = 2L * Nn;
    GC_zero_f32<<<(int)((n + 255) / 256), 256, 0, stream>>>(outdeg, n);
  }
  {
    long n = (long)Nn * OUT_FEATS;
    GC_zero_f32<<<(int)((n + 255) / 256), 256, 0, stream>>>(out, n);
  }

  // degree histograms
  GC_degrees<<<(E + 255) / 256, 256, 0, stream>>>(src, dst, outdeg, indeg, E);

  // one-time fp16 W transpose (L2-resident thereafter)
  GC_prepW<<<(IN_FEATS * OUT_FEATS + 255) / 256, 256, 0, stream>>>(weight, Wt);

  // WMMA GEMM + tanh
  int mblocks = (Nn + ROWS_PER_BLOCK - 1) / ROWS_PER_BLOCK;
  GC_gemm_tanh<<<mblocks, 256, 0, stream>>>(feat, Wt, outdeg, h, Nn);

  // edge scatter (L2-resident gathers + f32 atomics)
  GC_scatter<<<8192, 256, 0, stream>>>(h, ew, src, dst, out, E);

  // in-degree post-scale
  {
    long total = (long)Nn * OUT_FEATS;
    GC_scale_in<<<(int)((total + 255) / 256), 256, 0, stream>>>(out, indeg, total);
  }
}
